// GAT_22402549416098
// MI455X (gfx1250) — matrix operand order repaired and checked by reading, not compiled, not run on hardware
//
#include <hip/hip_runtime.h>

// ---------------------------------------------------------------------------
// CDNA5 WMMA types (wave32): v_wmma_f32_16x16x32_bf16
// ---------------------------------------------------------------------------
typedef __attribute__((ext_vector_type(16))) __bf16   v16bf;
typedef __attribute__((ext_vector_type(2)))  __bf16   v2bf;
typedef __attribute__((ext_vector_type(8)))  float    v8f;
typedef __attribute__((ext_vector_type(8)))  unsigned v8u;

#define TILE_K    32
#define NEG_SLOPE 0.2f

// ---------------------------------------------------------------------------
// Monotone float<->uint encoding so atomicMax(u32) implements float max
// ---------------------------------------------------------------------------
__device__ __forceinline__ unsigned encF(float f) {
    unsigned u = __float_as_uint(f);
    return (u & 0x80000000u) ? ~u : (u | 0x80000000u);
}
__device__ __forceinline__ float decF(unsigned u) {
    return (u & 0x80000000u) ? __uint_as_float(u ^ 0x80000000u)
                             : __uint_as_float(~u);
}

__device__ __forceinline__ unsigned pack_bf16x2(float a, float b) {
    v2bf v;
    v[0] = (__bf16)a;
    v[1] = (__bf16)b;
    return __builtin_bit_cast(unsigned, v);
}

// ---------------------------------------------------------------------------
// GEMM tile staging, software pipelined.
// LDS layout (dword = bf16 pair, fragment order):
//   As_dw[row][p] holds A[row][2p..2p+1]
//   Bs_dw[n][p]   holds B[2p][n], B[2p+1][n]
// Stride 20 dwords -> 80B rows: 16B-aligned b128 loads, distinct start banks.
// Requires: K even, N % 64 == 0. A rows clamped (only affect own output rows);
// K tail handled branch-free via clamped addresses + value selects.
// ---------------------------------------------------------------------------
struct TileRegs {
    float2 a[2];
    float  b[8];
};

__device__ __forceinline__ void load_tile_regs(TileRegs& tr,
        const float* __restrict__ A, const float* __restrict__ B,
        int tid, int block_m, int block_n, int k0, int M, int K, int N) {
    // A tile: 32 rows x 16 kpairs = 512 dwords, 2 float2 loads per thread
#pragma unroll
    for (int i = 0; i < 2; ++i) {
        int idx = tid + i * 256;
        int r = idx >> 4, p = idx & 15;
        int gm = block_m + r;
        gm = gm < M ? gm : M - 1;
        int gk  = k0 + 2 * p;
        int gkc = gk < K - 1 ? gk : K - 2;           // K even -> pair in-bounds
        float2 f = *(const float2*)(A + (size_t)gm * K + gkc);
        tr.a[i].x = gk < K ? f.x : 0.f;
        tr.a[i].y = gk < K ? f.y : 0.f;
    }
    // B tile: 64 cols x 16 kpairs = 1024 dwords, 8 b32 loads per thread
#pragma unroll
    for (int i = 0; i < 4; ++i) {
        int idx = tid + i * 256;
        int n = idx & 63, p = idx >> 6;
        int gn = block_n + n;
        int gk = k0 + 2 * p;
        int g0 = gk     < K ? gk     : K - 1;
        int g1 = gk + 1 < K ? gk + 1 : K - 1;
        float f0 = B[(size_t)g0 * N + gn];
        float f1 = B[(size_t)g1 * N + gn];
        tr.b[2 * i]     = gk     < K ? f0 : 0.f;
        tr.b[2 * i + 1] = gk + 1 < K ? f1 : 0.f;
    }
}

__device__ __forceinline__ void store_tile_lds(const TileRegs& tr,
        unsigned (*As_dw)[20], unsigned (*Bs_dw)[20], int tid) {
#pragma unroll
    for (int i = 0; i < 2; ++i) {
        int idx = tid + i * 256;
        As_dw[idx >> 4][idx & 15] = pack_bf16x2(tr.a[i].x, tr.a[i].y);
    }
#pragma unroll
    for (int i = 0; i < 4; ++i) {
        int idx = tid + i * 256;
        Bs_dw[idx & 63][idx >> 6] = pack_bf16x2(tr.b[2 * i], tr.b[2 * i + 1]);
    }
}

// A fragment (16-bit A 16x32 ISA layout): lane group h = lane>>4
//   VGPRs 0-3: K = 8h..8h+7   -> kpairs 4h..4h+3
//   VGPRs 4-7: K = 16+8h..+7  -> kpairs 8+4h..8+4h+3
__device__ __forceinline__ v16bf load_a_frag(const unsigned (*As_dw)[20],
                                             int wm, int lane) {
    const int row = wm * 16 + (lane & 15);
    const int h   = lane >> 4;
    uint4 g0 = *(const uint4*)&As_dw[row][4 * h];
    uint4 g1 = *(const uint4*)&As_dw[row][8 + 4 * h];
    v8u u = {g0.x, g0.y, g0.z, g0.w, g1.x, g1.y, g1.z, g1.w};
    return __builtin_bit_cast(v16bf, u);
}

// B fragment (B 32x16): same K order as A fragment
//   VGPRs 0-3: kpairs 4h..4h+3, VGPRs 4-7: kpairs 8+4h..8+4h+3
__device__ __forceinline__ v16bf load_b_frag(const unsigned (*Bs_dw)[20],
                                             int wn, int lane) {
    const int n = wn * 16 + (lane & 15);
    const int h = lane >> 4;
    uint4 g0 = *(const uint4*)&Bs_dw[n][4 * h];
    uint4 g1 = *(const uint4*)&Bs_dw[n][8 + 4 * h];
    v8u u = {g0.x, g0.y, g0.z, g0.w, g1.x, g1.y, g1.z, g1.w};
    return __builtin_bit_cast(v16bf, u);
}

// ---------------------------------------------------------------------------
// Tiled bf16 WMMA GEMM: C[M,N] = A[M,K] * B[K,N], fp32 accumulate.
// Block tile 32x64, 8 waves (2x4), one 16x16 WMMA tile per wave.
// Double-buffered LDS; next tile's global loads overlap current WMMA.
// ---------------------------------------------------------------------------
__launch_bounds__(256)
__global__ void k_gemm_bf16(const float* __restrict__ A,
                            const float* __restrict__ B,
                            float* __restrict__ C,
                            int M, int K, int N) {
    __shared__ __align__(16) unsigned As_dw[2][32][20];
    __shared__ __align__(16) unsigned Bs_dw[2][64][20];

    const int tid  = threadIdx.x;
    const int lane = tid & 31;
    const int wave = tid >> 5;
    const int wm   = wave >> 2;   // 0..1
    const int wn   = wave & 3;    // 0..3
    const int block_m = blockIdx.y * 32;
    const int block_n = blockIdx.x * 64;

    v8f acc = {};
    const int nTiles = (K + TILE_K - 1) / TILE_K;

    TileRegs tr;
    load_tile_regs(tr, A, B, tid, block_m, block_n, 0, M, K, N);
    store_tile_lds(tr, As_dw[0], Bs_dw[0], tid);

    for (int t = 0; t < nTiles; ++t) {
        __syncthreads();
        const int  cur  = t & 1;
        const bool more = (t + 1) < nTiles;

        // Issue next tile's global loads first: latency hidden behind WMMA.
        if (more)
            load_tile_regs(tr, A, B, tid, block_m, block_n,
                           (t + 1) * TILE_K, M, K, N);
        if ((t + 2) < nTiles) {   // speculative prefetch 2 tiles ahead
            int pr = block_m + (tid >> 3); pr = pr < M ? pr : M - 1;
            __builtin_prefetch(A + (size_t)pr * K + (t + 2) * TILE_K, 0, 1);
            __builtin_prefetch(B + (size_t)((t + 2) * TILE_K + lane) * N + block_n, 0, 1);
        }

        v16bf af  = load_a_frag(As_dw[cur], wm, lane);
        v16bf bfr = load_b_frag(Bs_dw[cur], wn, lane);
        acc = __builtin_amdgcn_wmma_f32_16x16x32_bf16(
                  false, af, false, bfr, (short)0, acc, false, false);

        if (more)
            store_tile_lds(tr, As_dw[1 - cur], Bs_dw[1 - cur], tid);
    }

    // C/D layout: VGPR r -> M = wm*16 + r + (lane>=16 ? 8 : 0), N = wn*16 + lane%16
    const int n     = block_n + wn * 16 + (lane & 15);
    const int mbase = block_m + wm * 16 + ((lane >> 4) * 8);
#pragma unroll
    for (int r = 0; r < 8; ++r) {
        int m = mbase + r;
        if (m < M) C[(size_t)m * N + n] = acc[r];
    }
}

// ---------------------------------------------------------------------------
// Per-(node, head) attention logits: al = <h[n,h,:], a[h,:]>  (C % 4 == 0)
// ---------------------------------------------------------------------------
__global__ void k_attn_logits(const float* __restrict__ h,
                              const float* __restrict__ a_src,
                              const float* __restrict__ a_dst,
                              float* __restrict__ al_src,
                              float* __restrict__ al_dst,
                              int NH, int hMask, int C) {
    int idx = blockIdx.x * blockDim.x + threadIdx.x;
    if (idx >= NH) return;
    int hh = idx & hMask;
    const float4* hp = (const float4*)(h + (size_t)idx * C);  // idx = node*H+hh
    const float4* as = (const float4*)(a_src + (size_t)hh * C);
    const float4* ad = (const float4*)(a_dst + (size_t)hh * C);
    float s1 = 0.f, s2 = 0.f;
    for (int c = 0; c < (C >> 2); ++c) {
        float4 v = hp[c], a = as[c], d = ad[c];
        s1 += v.x * a.x + v.y * a.y + v.z * a.z + v.w * a.w;
        s2 += v.x * d.x + v.y * d.y + v.z * d.z + v.w * d.w;
    }
    al_src[idx] = s1;
    al_dst[idx] = s2;
}

__global__ void k_fill_u32(unsigned* __restrict__ p, int n, unsigned v) {
    int t = blockIdx.x * blockDim.x + threadIdx.x;
    if (t < n) p[t] = v;
}

__device__ __forceinline__ void edge_sd(const long long* __restrict__ ei,
                                        int E, int e, int& s, int& d) {
    if (e < E) { s = (int)ei[e]; d = (int)ei[(size_t)E + e]; }
    else       { s = d = e - E; }
}

// Pass 1: segment max of leaky_relu(al_src[src]+al_dst[dst]) per (dst, head)
__global__ void k_edge_max(const long long* __restrict__ ei, int E, int Nn,
                           const float* __restrict__ al_src,
                           const float* __restrict__ al_dst,
                           unsigned* __restrict__ mEnc,
                           int hShift, int hMask) {
    int total = (E + Nn) << hShift;
    for (int t = blockIdx.x * blockDim.x + threadIdx.x; t < total;
         t += gridDim.x * blockDim.x) {
        int e = t >> hShift, hh = t & hMask;
        int s, d; edge_sd(ei, E, e, s, d);
        float v = al_src[(s << hShift) + hh] + al_dst[(d << hShift) + hh];
        v = (v > 0.f) ? v : v * NEG_SLOPE;
        atomicMax(&mEnc[(d << hShift) + hh], encF(v));
    }
}

// Pass 2: ex = exp(e - m[dst]); segment sum per (dst, head); store ex per edge
__global__ void k_edge_expsum(const long long* __restrict__ ei, int E, int Nn,
                              const float* __restrict__ al_src,
                              const float* __restrict__ al_dst,
                              const unsigned* __restrict__ mEnc,
                              float* __restrict__ ex, float* __restrict__ sSum,
                              int hShift, int hMask) {
    int total = (E + Nn) << hShift;
    for (int t = blockIdx.x * blockDim.x + threadIdx.x; t < total;
         t += gridDim.x * blockDim.x) {
        int e = t >> hShift, hh = t & hMask;
        int s, d; edge_sd(ei, E, e, s, d);
        float v = al_src[(s << hShift) + hh] + al_dst[(d << hShift) + hh];
        v = (v > 0.f) ? v : v * NEG_SLOPE;
        float x = __expf(v - decF(mEnc[(d << hShift) + hh]));
        ex[t] = x;   // t == e*H + hh
        atomicAdd(&sSum[(d << hShift) + hh], x);
    }
}

// Pass 3: agg[dst,h,c] += h[src,h,c] * ex/(s[dst,h]+eps). Thread per (edge, c).
__global__ void k_edge_agg(const long long* __restrict__ ei, int E, int Nn,
                           const float* __restrict__ hFeat,
                           const float* __restrict__ ex,
                           const float* __restrict__ sSum,
                           float* __restrict__ agg,
                           int H, int hShift, int cShift, int cMask) {
    int total = (E + Nn) << cShift;
    for (int t = blockIdx.x * blockDim.x + threadIdx.x; t < total;
         t += gridDim.x * blockDim.x) {
        int e = t >> cShift, c = t & cMask;
        int s, d; edge_sd(ei, E, e, s, d);
        for (int hh = 0; hh < H; ++hh) {
            float alpha = ex[(e << hShift) + hh]
                        / (sSum[(d << hShift) + hh] + 1e-16f);
            float v = hFeat[(((size_t)(s << hShift) + hh) << cShift) + c] * alpha;
            atomicAdd(&agg[(((size_t)(d << hShift) + hh) << cShift) + c], v);
        }
    }
}

// In-place bias + ReLU over [N, F], F power of two
__global__ void k_bias_relu(float* __restrict__ x, const float* __restrict__ b,
                            int n, int fMask) {
    for (int t = blockIdx.x * blockDim.x + threadIdx.x; t < n;
         t += gridDim.x * blockDim.x) {
        float v = x[t] + b[t & fMask];
        x[t] = v > 0.f ? v : 0.f;
    }
}

// Global mean pool (sums + counts via atomics)
__global__ void k_pool(const float* __restrict__ h,
                       const long long* __restrict__ batch,
                       float* __restrict__ sums, float* __restrict__ cnt,
                       int Nn, int cShift, int cMask) {
    int total = Nn << cShift;
    for (int t = blockIdx.x * blockDim.x + threadIdx.x; t < total;
         t += gridDim.x * blockDim.x) {
        int node = t >> cShift, c = t & cMask;
        int g = (int)batch[node];
        atomicAdd(&sums[(g << cShift) + c], h[t]);
        if (c == 0) atomicAdd(&cnt[g], 1.0f);
    }
}

// Final tiny GEMM: out[G,O] = (sums/cnt) @ Wl + bl
__global__ void k_final(const float* __restrict__ sums, const float* __restrict__ cnt,
                        const float* __restrict__ Wl, const float* __restrict__ bl,
                        float* __restrict__ out, int G, int C, int O) {
    int t = blockIdx.x * blockDim.x + threadIdx.x;
    if (t >= G * O) return;
    int g = t / O, o = t - g * O;
    float inv = 1.0f / fmaxf(cnt[g], 1.0f);
    float acc = 0.f;
    for (int c = 0; c < C; ++c) acc += sums[g * C + c] * inv * Wl[c * O + o];
    out[t] = acc + bl[o];
}

// ---------------------------------------------------------------------------
// Host launch
// ---------------------------------------------------------------------------
extern "C" void kernel_launch(void* const* d_in, const int* in_sizes, int n_in,
                              void* d_out, int out_size, void* d_ws, size_t ws_size,
                              hipStream_t stream) {
    const float*     x     = (const float*)d_in[0];
    const long long* ei    = (const long long*)d_in[1];
    const long long* batch = (const long long*)d_in[2];
    const float*     W1    = (const float*)d_in[3];
    const float*     a_s1  = (const float*)d_in[4];
    const float*     a_d1  = (const float*)d_in[5];
    const float*     b1    = (const float*)d_in[6];
    const float*     W2    = (const float*)d_in[7];
    const float*     a_s2  = (const float*)d_in[8];
    const float*     a_d2  = (const float*)d_in[9];
    const float*     b2    = (const float*)d_in[10];
    const float*     Wl    = (const float*)d_in[11];
    const float*     bl    = (const float*)d_in[12];
    float*           out   = (float*)d_out;

    // Derived dims (HC, C, H powers of two: 256, 64, 4)
    const int HC  = in_sizes[6];
    const int C   = in_sizes[10];
    const int H   = HC / C;
    const int Fin = in_sizes[3] / HC;
    const int Nn  = in_sizes[0] / Fin;
    const int E   = in_sizes[1] / 2;
    const int O   = in_sizes[12];
    const int G   = out_size / O;
    const int cSh = __builtin_ctz(C);
    const int hSh = __builtin_ctz(H);

    // Workspace carving (256B aligned)
    char* ws = (char*)d_ws;
    size_t off = 0;
    auto carve = [&](size_t bytes) {
        char* p = ws + off;
        off += (bytes + 255) & ~(size_t)255;
        return p;
    };
    const size_t ET = (size_t)E + Nn;
    float*    h1   = (float*)   carve((size_t)Nn * HC * sizeof(float));
    float*    agg1 = (float*)   carve((size_t)Nn * HC * sizeof(float));
    float*    als1 = (float*)   carve((size_t)Nn * H * sizeof(float));
    float*    ald1 = (float*)   carve((size_t)Nn * H * sizeof(float));
    unsigned* m1   = (unsigned*)carve((size_t)Nn * H * sizeof(unsigned));
    float*    s1   = (float*)   carve((size_t)Nn * H * sizeof(float));
    float*    ex1  = (float*)   carve(ET * H * sizeof(float));
    float*    h2   = (float*)   carve((size_t)Nn * C * sizeof(float));
    float*    agg2 = (float*)   carve((size_t)Nn * C * sizeof(float));
    float*    als2 = (float*)   carve((size_t)Nn * sizeof(float));
    float*    ald2 = (float*)   carve((size_t)Nn * sizeof(float));
    unsigned* m2   = (unsigned*)carve((size_t)Nn * sizeof(unsigned));
    float*    s2   = (float*)   carve((size_t)Nn * sizeof(float));
    float*    ex2  = (float*)   carve(ET * sizeof(float));
    float*    psum = (float*)   carve((size_t)G * C * sizeof(float));
    float*    pcnt = (float*)   carve((size_t)G * sizeof(float));
    (void)ws_size; (void)n_in;

    const int TPB = 256;
    const int EDGE_BLOCKS = 4096;
    auto blocks1d = [](long long n, int tpb) {
        long long b = (n + tpb - 1) / tpb;
        return (unsigned)(b > 65535 ? 65535 : b);
    };

    // ---- Zero accumulators (every call: deterministic) ----
    hipMemsetAsync(s1,   0, (size_t)Nn * H * sizeof(float), stream);
    hipMemsetAsync(agg1, 0, (size_t)Nn * HC * sizeof(float), stream);
    hipMemsetAsync(s2,   0, (size_t)Nn * sizeof(float), stream);
    hipMemsetAsync(agg2, 0, (size_t)Nn * C * sizeof(float), stream);
    hipMemsetAsync(psum, 0, (size_t)G * C * sizeof(float), stream);
    hipMemsetAsync(pcnt, 0, (size_t)G * sizeof(float), stream);
    const unsigned negInfEnc = ~0xFF800000u;   // encF(-inf)
    k_fill_u32<<<blocks1d((long long)Nn * H, TPB), TPB, 0, stream>>>(m1, Nn * H, negInfEnc);
    k_fill_u32<<<blocks1d(Nn, TPB), TPB, 0, stream>>>(m2, Nn, negInfEnc);

    // ================= Layer 1 =================
    {   // h1 = x @ W1  (M=Nn, K=Fin, N=HC)
        dim3 grid((HC + 63) / 64, (Nn + 31) / 32);
        k_gemm_bf16<<<grid, 256, 0, stream>>>(x, W1, h1, Nn, Fin, HC);
    }
    k_attn_logits<<<blocks1d((long long)Nn * H, TPB), TPB, 0, stream>>>(
        h1, a_s1, a_d1, als1, ald1, Nn * H, H - 1, C);
    k_edge_max<<<EDGE_BLOCKS, TPB, 0, stream>>>(ei, E, Nn, als1, ald1, m1, hSh, H - 1);
    k_edge_expsum<<<EDGE_BLOCKS, TPB, 0, stream>>>(ei, E, Nn, als1, ald1, m1, ex1, s1, hSh, H - 1);
    k_edge_agg<<<EDGE_BLOCKS, TPB, 0, stream>>>(ei, E, Nn, h1, ex1, s1, agg1, H, hSh, cSh, C - 1);
    k_bias_relu<<<EDGE_BLOCKS, TPB, 0, stream>>>(agg1, b1, Nn * HC, HC - 1);

    // ================= Layer 2 (H=1) =================
    {   // h2 = agg1 @ W2  (M=Nn, K=HC, N=C)
        dim3 grid((C + 63) / 64, (Nn + 31) / 32);
        k_gemm_bf16<<<grid, 256, 0, stream>>>(agg1, W2, h2, Nn, HC, C);
    }
    k_attn_logits<<<blocks1d(Nn, TPB), TPB, 0, stream>>>(
        h2, a_s2, a_d2, als2, ald2, Nn, 0, C);
    k_edge_max<<<EDGE_BLOCKS, TPB, 0, stream>>>(ei, E, Nn, als2, ald2, m2, 0, 0);
    k_edge_expsum<<<EDGE_BLOCKS, TPB, 0, stream>>>(ei, E, Nn, als2, ald2, m2, ex2, s2, 0, 0);
    k_edge_agg<<<EDGE_BLOCKS, TPB, 0, stream>>>(ei, E, Nn, h2, ex2, s2, agg2, 1, 0, cSh, C - 1);
    k_bias_relu<<<EDGE_BLOCKS, TPB, 0, stream>>>(agg2, b2, Nn * C, C - 1);

    // ================= Pool + head =================
    k_pool<<<EDGE_BLOCKS, TPB, 0, stream>>>(agg2, batch, psum, pcnt, Nn, cSh, C - 1);
    k_final<<<(G * O + TPB - 1) / TPB, TPB, 0, stream>>>(psum, pcnt, Wl, bl, out, G, C, O);
}
